// VQDecoderV2_84155589198324
// MI455X (gfx1250) — matrix operand order repaired and checked by reading, not compiled, not run on hardware
//
#include <hip/hip_runtime.h>

// ---------------------------------------------------------------------------
// CDNA5 (gfx1250, wave32) implicit-GEMM conv1d decoder for the VQDecoder net.
// v_wmma_f32_16x16x32_bf16, double-buffered LDS pipeline (1 barrier / k-step),
// async global->LDS copy for the (pure-bf16) weight tile where available.
// ---------------------------------------------------------------------------

typedef __attribute__((ext_vector_type(16))) __bf16   v16bf;
typedef __attribute__((ext_vector_type(8)))  float    v8f;
typedef __attribute__((ext_vector_type(4)))  unsigned v4u;
typedef __attribute__((ext_vector_type(4)))  int      v4i;

#if __has_builtin(__builtin_amdgcn_global_load_async_to_lds_b128)
#define HAVE_ASYNC_LDS 1
#endif

__device__ __forceinline__ void wait_asynccnt0() {
#if __has_builtin(__builtin_amdgcn_s_wait_asynccnt)
  __builtin_amdgcn_s_wait_asynccnt(0);
#else
  asm volatile("s_wait_asynccnt 0x0" ::: "memory");
#endif
}

__device__ __forceinline__ unsigned short f2bf(float f) {
  union { float f; unsigned u; } v; v.f = f;
  unsigned r = v.u + 0x7fffu + ((v.u >> 16) & 1u);   // round-to-nearest-even
  return (unsigned short)(r >> 16);
}
__device__ __forceinline__ unsigned pack2bf(float lo, float hi) {
#if __has_builtin(__builtin_amdgcn_cvt_pk_bf16_f32)
  typedef __attribute__((ext_vector_type(2))) __bf16 v2bf;
  union { v2bf v; unsigned u; } c;
  c.v = __builtin_amdgcn_cvt_pk_bf16_f32(lo, hi);    // v_cvt_pk_bf16_f32
  return c.u;
#else
  return (unsigned)f2bf(lo) | ((unsigned)f2bf(hi) << 16);
#endif
}

union Frag { v16bf v; unsigned u[8]; };

#define BM 64
#define BN 128
#define BK 32
#define AS 40    // A LDS row stride (ushorts): 80B rows -> b128-aligned stores, conflict-free b32 reads
#define BSTR 132 // B LDS pair-row stride (uints): 528B rows -> b128-aligned stores, conflict-free reads

// Y[n,o,w] = act( sum_{ci,tap} W[o,ci,tap] * Xeff[n,ci, w*STRIDE+tap-1] + bias[o] ) (+ Res)
// Xeff = UP2 ? repeat2(X) : X.  Wp is bf16 packed+padded [tap][kb][Opad][32].
template<int STRIDE, bool UP2, bool LRELU, bool RES, bool TSTORE>
__global__ __launch_bounds__(256)
void conv1d_wmma(const float* __restrict__ X,
                 const unsigned short* __restrict__ Wp,
                 const float* __restrict__ Bias,
                 const float* __restrict__ Res,
                 float* __restrict__ Y,
                 int Cin, int Win, int O, int Wout, int Opad)
{
  __shared__ __align__(16) unsigned short Alds[2][BM * AS];
  __shared__ __align__(16) unsigned       Bp[2][(BK / 2) * BSTR];

  const int n    = blockIdx.z;
  const int ob   = blockIdx.y * BM;
  const int wb   = blockIdx.x * BN;
  const int tid  = threadIdx.x;
  const int lane = tid & 31;
  const int wave = tid >> 5;
  const int lid  = lane & 15;
  const int hi   = lane >> 4;
  const int mw   = (wave & 3) * 16;   // wave M offset
  const int nw   = (wave >> 2) * 64;  // wave N offset (4 accumulators of 16)

  const int WinEff = UP2 ? (Win * 2) : Win;
  const float* Xn = X + (size_t)n * Cin * Win;

  const v8f zero8 = {0.f,0.f,0.f,0.f,0.f,0.f,0.f,0.f};
  v8f acc0 = zero8, acc1 = zero8, acc2 = zero8, acc3 = zero8;

  // staging decompositions (256 threads)
  const int arow = tid >> 1;           // A: 2 threads per row, 16B each
  const int ac0  = (tid & 1) * 16;
  const int rp   = tid >> 4;           // B: K-pair row 0..15
  const int cg   = (tid & 15) * 8;     // B: 8 columns per thread

  const int kblocks = (Cin + BK - 1) / BK;
  const int ktotal  = 3 * kblocks;

  // ---- pipeline registers ----
#ifndef HAVE_ASYNC_LDS
  v4u aReg;
#endif
  unsigned bReg[8];

  // A tile: pure bf16 copy global->LDS (async DMA path when available)
  auto load_A = [&](int tap, int kb, int buf) {
    const unsigned short* src =
        Wp + (((size_t)tap * kblocks + kb) * Opad + ob) * BK + arow * BK + ac0;
#ifdef HAVE_ASYNC_LDS
    unsigned short* dst = &Alds[buf][arow * AS + ac0];
    __builtin_amdgcn_global_load_async_to_lds_b128((v4i*)src, (v4i*)dst, 0, 0);
#else
    (void)buf;
    aReg = *(const v4u*)src;
#endif
  };
  auto load_B = [&](int tap, int kb) {
    const int ci0 = kb * BK + 2 * rp;
    const bool interior = ((wb * STRIDE + tap - 1) >= 0) &&
                          (((wb + BN - 1) * STRIDE + tap - 1) < WinEff) &&
                          ((kb * BK + BK) <= Cin);
    if (interior) {
      const float* r0 = Xn + (size_t)ci0 * Win;
      const float* r1 = r0 + Win;
      #pragma unroll
      for (int j = 0; j < 8; ++j) {
        const int wi = (wb + cg + j) * STRIDE + tap - 1;
        const int ws = UP2 ? (wi >> 1) : wi;
        bReg[j] = pack2bf(r0[ws], r1[ws]);
      }
      __builtin_prefetch(r0 + (size_t)BK * Win, 0, 1);  // next K-slab (speculative)
    } else {
      #pragma unroll
      for (int j = 0; j < 8; ++j) {
        const int wi = (wb + cg + j) * STRIDE + tap - 1;
        const int ws = UP2 ? (wi >> 1) : wi;
        float a = 0.f, b = 0.f;
        if (wi >= 0 && wi < WinEff) {
          if (ci0 < Cin)     a = Xn[(size_t)ci0 * Win + ws];
          if (ci0 + 1 < Cin) b = Xn[((size_t)ci0 + 1) * Win + ws];
        }
        bReg[j] = pack2bf(a, b);
      }
    }
  };
  auto store_tile = [&](int buf) {
#ifndef HAVE_ASYNC_LDS
    *(v4u*)&Alds[buf][arow * AS + ac0] = aReg;
#endif
    v4u o0 = {bReg[0], bReg[1], bReg[2], bReg[3]};
    v4u o1 = {bReg[4], bReg[5], bReg[6], bReg[7]};
    *(v4u*)&Bp[buf][rp * BSTR + cg]     = o0;
    *(v4u*)&Bp[buf][rp * BSTR + cg + 4] = o1;
  };

  // ---- prologue: fill buffer 0 ----
  load_A(0, 0, 0);
  load_B(0, 0);
  store_tile(0);
#ifdef HAVE_ASYNC_LDS
  wait_asynccnt0();
#endif
  __syncthreads();

  int tapN = (kblocks == 1) ? 1 : 0;      // indices of tile kk+1
  int kbN  = (kblocks == 1) ? 0 : 1;

  for (int kk = 0; kk < ktotal; ++kk) {
    const int cur = kk & 1;
    const bool more = (kk + 1) < ktotal;
    if (more) {
      load_A(tapN, kbN, cur ^ 1);         // async DMA straight into next buffer
      load_B(tapN, kbN);                  // global loads overlap WMMA below
    }

    // ---- A fragment (ISA 7.12.2, 16-bit A 16x32): merged ds_load_b128 ----
    Frag af;
    {
      const int m = mw + lid;
      #pragma unroll
      for (int j = 0; j < 8; ++j) {
        const int k0 = (j < 4) ? (hi * 8 + 2 * j) : (16 + hi * 8 + 2 * (j - 4));
        af.u[j] = *(const unsigned*)&Alds[cur][m * AS + k0];
      }
    }
    // ---- 4 B fragments, reuse A across 4 WMMAs ----
    #pragma unroll
    for (int h = 0; h < 4; ++h) {
      Frag bf;
      const int col = nw + h * 16 + lid;
      #pragma unroll
      for (int j = 0; j < 8; ++j)
        bf.u[j] = Bp[cur][((j < 4) ? (hi * 4 + j) : (8 + hi * 4 + (j - 4))) * BSTR + col];
      v8f& a = (h == 0) ? acc0 : (h == 1) ? acc1 : (h == 2) ? acc2 : acc3;
      a = __builtin_amdgcn_wmma_f32_16x16x32_bf16(false, af.v, false, bf.v,
                                                  (short)0, a, false, false);
    }

    if (more) {
      store_tile(cur ^ 1);
      kbN++;
      if (kbN == kblocks) { kbN = 0; tapN++; }
#ifdef HAVE_ASYNC_LDS
      wait_asynccnt0();                   // async A-tile landed before the barrier
#endif
    }
    __syncthreads();
  }

  // ---- epilogue: bias -> lrelu -> +residual; C/D layout: VGPR r -> M = r + hi*8 ----
  float bias8[8];
  #pragma unroll
  for (int r = 0; r < 8; ++r) {
    const int o = ob + mw + hi * 8 + r;
    bias8[r] = (o < O) ? Bias[o] : 0.f;
  }
  const bool fullM = (ob + BM) <= O;
  #pragma unroll
  for (int h = 0; h < 4; ++h) {
    const v8f a = (h == 0) ? acc0 : (h == 1) ? acc1 : (h == 2) ? acc2 : acc3;
    const int w = wb + nw + h * 16 + lid;
    #pragma unroll
    for (int r = 0; r < 8; ++r) {
      const int o = ob + mw + hi * 8 + r;
      if ((fullM || o < O) && w < Wout) {
        float v = a[r] + bias8[r];
        if (LRELU) v = (v >= 0.f) ? v : 0.2f * v;
        if (RES)   v += Res[((size_t)n * O + o) * (size_t)Wout + w];
        const size_t idx = TSTORE ? (((size_t)n * Wout + w) * (size_t)O + o)
                                  : (((size_t)n * O + o) * (size_t)Wout + w);
        Y[idx] = v;
      }
    }
  }
}

// (N,W,C) fp32 -> (N,C,W) fp32, N=32
__global__ void transpose_in_kernel(const float* __restrict__ in, float* __restrict__ out,
                                    int Wd, int C) {
  size_t i = (size_t)blockIdx.x * blockDim.x + threadIdx.x;
  const size_t total = (size_t)32 * Wd * C;
  if (i >= total) return;
  const int c = (int)(i % C);
  const size_t r = i / C;
  const int w = (int)(r % Wd);
  const int n = (int)(r / Wd);
  out[((size_t)n * C + c) * (size_t)Wd + w] = in[i];
}

// W[o,ci,k] fp32 -> Wp[tap][kb][Opad][32] bf16, zero-padded
__global__ void pack_w_kernel(const float* __restrict__ w, unsigned short* __restrict__ wp,
                              int O, int Cin, int Opad, int kblocks, size_t total) {
  size_t i = (size_t)blockIdx.x * blockDim.x + threadIdx.x;
  if (i >= total) return;
  const int ci32 = (int)(i & 31);
  size_t r = i >> 5;
  const int o = (int)(r % Opad);
  r /= Opad;
  const int kb  = (int)(r % kblocks);
  const int tap = (int)(r / kblocks);
  const int ci = kb * 32 + ci32;
  unsigned short v = 0;
  if (o < O && ci < Cin) v = f2bf(w[((size_t)o * Cin + ci) * 3 + tap]);
  wp[i] = v;
}

// cat[n, c, w] = fin[n, c, w] (c<75);  cat[n, 75+c, w] = fin[n, c, w] (c>=3)
__global__ void build_cat_kernel(const float* __restrict__ fin, float* __restrict__ cat,
                                 int W, size_t total) {
  size_t i = (size_t)blockIdx.x * blockDim.x + threadIdx.x;
  if (i >= total) return;
  const int w = (int)(i % W);
  const size_t r = i / W;
  const int c = (int)(r % 75);
  const int n = (int)(r / 75);
  const float v = fin[i];
  float* base = cat + (size_t)n * 150 * W;
  base[(size_t)c * W + w] = v;
  if (c >= 3) base[(size_t)(75 + c) * W + w] = v;
}

// cat[n, 75+c, w] = inclusive cumsum over w of fin[n, c, :] with element 0 zeroed (c<3)
__global__ void scan3_kernel(const float* __restrict__ fin, float* __restrict__ cat, int W) {
  const int n = blockIdx.x, c = blockIdx.y;
  const float* src = fin + ((size_t)n * 75 + c) * (size_t)W;
  float* dst = cat + ((size_t)n * 150 + 75 + c) * (size_t)W;
  __shared__ float part[256];
  const int t = threadIdx.x;
  const int per = W / 256;   // 8 for W=2048
  float loc[16];
  float run = 0.f;
  for (int j = 0; j < per; ++j) {
    const int w = t * per + j;
    const float v = (w == 0) ? 0.f : src[w];
    run += v; loc[j] = run;
  }
  part[t] = run;
  __syncthreads();
  for (int off = 1; off < 256; off <<= 1) {   // Hillis-Steele inclusive scan
    const float v = (t >= off) ? part[t - off] : 0.f;
    __syncthreads();
    part[t] += v;
    __syncthreads();
  }
  const float excl = (t > 0) ? part[t - 1] : 0.f;
  for (int j = 0; j < per; ++j) dst[t * per + j] = loc[j] + excl;
}

// ---------------------------------------------------------------------------

static inline void launch_conv(hipStream_t s, const float* X, const unsigned short* Wp,
                               const float* bias, const float* res, float* Y,
                               int Cin, int Win, int O, int Wout,
                               int stride, bool up2, bool lrelu, bool tstore) {
  const int Opad = ((O + 63) / 64) * 64;
  dim3 g((Wout + BN - 1) / BN, (O + 63) / 64, 32), b(256);
  const bool rs = (res != nullptr);
  if (tstore)          conv1d_wmma<1,false,false,false,true ><<<g,b,0,s>>>(X,Wp,bias,res,Y,Cin,Win,O,Wout,Opad);
  else if (stride==2)  conv1d_wmma<2,false,true ,false,false><<<g,b,0,s>>>(X,Wp,bias,res,Y,Cin,Win,O,Wout,Opad);
  else if (up2 && rs)  conv1d_wmma<1,true ,true ,true ,false><<<g,b,0,s>>>(X,Wp,bias,res,Y,Cin,Win,O,Wout,Opad);
  else if (up2)        conv1d_wmma<1,true ,true ,false,false><<<g,b,0,s>>>(X,Wp,bias,res,Y,Cin,Win,O,Wout,Opad);
  else if (rs)         conv1d_wmma<1,false,false,true ,false><<<g,b,0,s>>>(X,Wp,bias,res,Y,Cin,Win,O,Wout,Opad);
  else if (lrelu)      conv1d_wmma<1,false,true ,false,false><<<g,b,0,s>>>(X,Wp,bias,res,Y,Cin,Win,O,Wout,Opad);
  else                 conv1d_wmma<1,false,false,false,false><<<g,b,0,s>>>(X,Wp,bias,res,Y,Cin,Win,O,Wout,Opad);
}

extern "C" void kernel_launch(void* const* d_in, const int* in_sizes, int n_in,
                              void* d_out, int out_size, void* d_ws, size_t ws_size,
                              hipStream_t stream) {
  (void)in_sizes; (void)n_in; (void)out_size; (void)ws_size;
  char* ws = (char*)d_ws;
  const float* x_in = (const float*)d_in[0];

  // jax tree-flatten order (dicts sorted by key): inputs, c0, final, res[0..2]{c1,c2},
  // unet{d1,d1r{c1,c2},d2,d2r{c1,c2},inp,out,u1,u1r{c1,c2},u2}, up0, up1. Each param = (w, b).
  struct Layer { int wi; int O; int Cin; size_t off; };
  Layer Ls[22] = {
    { 1,1024, 512,0},                                        // 0  c0
    { 5,1024,1024,0},{ 7,1024,1024,0},                       // 1,2  res0 c1/c2
    { 9,1024,1024,0},{11,1024,1024,0},                       // 3,4  res1
    {13,1024,1024,0},{15,1024,1024,0},                       // 5,6  res2
    {41,1024,1024,0},                                        // 7  up0
    {43,  75,1024,0},                                        // 8  up1
    { 3,  75,  75,0},                                        // 9  final
    {29, 256, 150,0},                                        // 10 unet.inp
    {17, 512, 256,0},                                        // 11 d1
    {19, 512, 512,0},{21, 512, 512,0},                       // 12,13 d1r
    {23,1024, 512,0},                                        // 14 d2
    {25,1024,1024,0},{27,1024,1024,0},                       // 15,16 d2r
    {33, 512,1024,0},                                        // 17 u1
    {35, 512, 512,0},{37, 512, 512,0},                       // 18,19 u1r
    {39, 256, 512,0},                                        // 20 u2
    {31,  75, 256,0}                                         // 21 unet.out
  };
  size_t woff = 0;
  for (int i = 0; i < 22; ++i) {
    Ls[i].off = woff;
    const int Opad = ((Ls[i].O + 63) / 64) * 64;
    const int kb   = (Ls[i].Cin + 31) / 32;
    size_t bytes = (size_t)3 * kb * Opad * 32 * sizeof(unsigned short);
    woff += (bytes + 255) & ~(size_t)255;
  }
  const size_t MiB = 1ull << 20;
  const size_t R1 = (woff + 255) & ~(size_t)255;  // 64 MiB region
  const size_t R2 = R1 + 64 * MiB;                // 64 MiB
  const size_t R3 = R2 + 64 * MiB;                // 128 MiB (two 64 MiB halves)
  const size_t R4 = R3 + 128 * MiB;               // 64 MiB  -> total ~392 MiB

  // pack all weights to bf16 [tap][kb][Opad][32] (zero-padded)
  for (int i = 0; i < 22; ++i) {
    const int Opad = ((Ls[i].O + 63) / 64) * 64;
    const int kb   = (Ls[i].Cin + 31) / 32;
    size_t tot = (size_t)3 * kb * Opad * 32;
    pack_w_kernel<<<(unsigned)((tot + 255) / 256), 256, 0, stream>>>(
        (const float*)d_in[Ls[i].wi], (unsigned short*)(ws + Ls[i].off),
        Ls[i].O, Ls[i].Cin, Opad, kb, tot);
  }
  auto WPK = [&](int li) { return (const unsigned short*)(ws + Ls[li].off); };
  auto BIA = [&](int li) { return (const float*)d_in[Ls[li].wi + 1]; };

  float* bR1  = (float*)(ws + R1);
  float* bR2  = (float*)(ws + R2);
  float* bR3  = (float*)(ws + R3);
  float* bR3b = (float*)(ws + R3 + 64 * MiB);
  float* bR4  = (float*)(ws + R4);

  // 0) (N,W,C) -> (N,C,W)
  { size_t tot = (size_t)32 * 512 * 512;
    transpose_in_kernel<<<(unsigned)((tot + 255) / 256), 256, 0, stream>>>(x_in, bR1, 512, 512); }

  // 1) c0: 512 -> 1024, W=512
  launch_conv(stream, bR1, WPK(0), BIA(0), nullptr, bR2, 512, 512, 1024, 512, 1, false, false, false);

  // 2) 3 resblocks at 1024ch, W=512 (ping-pong R2 <-> R1, tmp in R3)
  float* x = bR2; float* other = bR1;
  for (int i = 0; i < 3; ++i) {
    launch_conv(stream, x,   WPK(1+2*i), BIA(1+2*i), nullptr, bR3,   1024, 512, 1024, 512, 1, false, true,  false);
    launch_conv(stream, bR3, WPK(2+2*i), BIA(2+2*i), x,       other, 1024, 512, 1024, 512, 1, false, false, false);
    float* t = x; x = other; other = t;
  }
  // x == bR1 now
  // 3) up0: up2 + conv(1024->1024) + lrelu, W: 512 -> 1024
  launch_conv(stream, x, WPK(7), BIA(7), nullptr, bR3, 1024, 512, 1024, 1024, 1, true, true, false);
  // 4) up1: up2 + conv(1024->75) + lrelu, W: 1024 -> 2048
  float* F75a = bR2;
  launch_conv(stream, bR3, WPK(8), BIA(8), nullptr, F75a, 1024, 1024, 75, 2048, 1, true, true, false);
  // 5) final: 75 -> 75, W=2048
  float* F75b = (float*)(ws + R2 + 24 * MiB);
  launch_conv(stream, F75a, WPK(9), BIA(9), nullptr, F75b, 75, 2048, 75, 2048, 1, false, false, false);
  // 6) cat (150ch, NCW): copy + 3-channel prefix sums
  float* CAT = bR1;
  { size_t tot = (size_t)32 * 75 * 2048;
    build_cat_kernel<<<(unsigned)((tot + 255) / 256), 256, 0, stream>>>(F75b, CAT, 2048, tot); }
  { dim3 g(32, 3); scan3_kernel<<<g, 256, 0, stream>>>(F75b, CAT, 2048); }

  // 7) unet.inp: 150 -> 256, W=2048
  float* x0 = bR2;
  launch_conv(stream, CAT, WPK(10), BIA(10), nullptr, x0, 150, 2048, 256, 2048, 1, false, false, false);
  // 8) d1 (stride2 + lrelu) + d1r resblock
  launch_conv(stream, x0,   WPK(11), BIA(11), nullptr, bR3,  256, 2048, 512, 1024, 2, false, true,  false);
  launch_conv(stream, bR3,  WPK(12), BIA(12), nullptr, bR3b, 512, 1024, 512, 1024, 1, false, true,  false);
  float* x1 = bR4;
  launch_conv(stream, bR3b, WPK(13), BIA(13), bR3,    x1,   512, 1024, 512, 1024, 1, false, false, false);
  // 9) d2 (stride2 + lrelu) + d2r resblock
  launch_conv(stream, x1,   WPK(14), BIA(14), nullptr, bR3,  512, 1024, 1024, 512, 2, false, true,  false);
  launch_conv(stream, bR3,  WPK(15), BIA(15), nullptr, bR3b, 1024, 512, 1024, 512, 1, false, true,  false);
  float* x2 = bR1;  // CAT dead
  launch_conv(stream, bR3b, WPK(16), BIA(16), bR3,    x2,   1024, 512, 1024, 512, 1, false, false, false);
  // 10) u1: up2 + conv(1024->512) + lrelu + x1 residual, W: 512 -> 1024
  launch_conv(stream, x2, WPK(17), BIA(17), x1, bR3, 1024, 512, 512, 1024, 1, true, true, false);
  // 11) u1r resblock
  launch_conv(stream, bR3,  WPK(18), BIA(18), nullptr, bR3b, 512, 1024, 512, 1024, 1, false, true,  false);
  float* y2 = bR4;  // x1 dead
  launch_conv(stream, bR3b, WPK(19), BIA(19), bR3,    y2,   512, 1024, 512, 1024, 1, false, false, false);
  // 12) u2: up2 + conv(512->256) + lrelu + x0 residual, W: 1024 -> 2048
  float* y1 = bR1;  // x2 dead
  launch_conv(stream, y2, WPK(20), BIA(20), x0, y1, 512, 1024, 256, 2048, 1, true, true, false);
  // 13) unet.out: 256 -> 75, W=2048, store transposed (N,W,C) directly into d_out
  launch_conv(stream, y1, WPK(21), BIA(21), nullptr, (float*)d_out, 256, 2048, 75, 2048, 1, false, false, true);
}
